// HawkesProcessDensity_81063212745631
// MI455X (gfx1250) — compile-verified, matrix operand order checked by reading, not run on hardware
//
#include <hip/hip_runtime.h>

typedef __attribute__((ext_vector_type(16))) _Float16 v16h;
typedef __attribute__((ext_vector_type(8)))  float    v8f;

#define N_EVENTS_MAX 8192
#define TILE_M 16
#define TILE_K 32
#define WAVES_PER_BLOCK 8
#define T_HORIZON 50.0f
#define TRAP_STEP 1e-5f
#define N_GRID 5000000
#define INT_BLOCKS 1024
#define INT_THREADS 256

// One 16x32 K-chunk: 4x ds_load_b128 of column times, 16 branchless exps,
// one v_wmma_f32_16x16x32_f16 row-reduction against an all-ones B.
template <bool MASKED>
__device__ __forceinline__ v8f chunk_step(const float4* __restrict__ sT4,
                                          int kk, int g, int row, float ti,
                                          float bs, v16h bones, v8f acc) {
  const int kbase = kk + (g << 3);            // first K this lane holds
  const int f4    = (kk >> 2) + (g << 1);     // float4 index of that K
  const float4 c0 = sT4[f4];                  // K = kbase + 0..3
  const float4 c1 = sT4[f4 + 1];              // K = kbase + 4..7
  const float4 c2 = sT4[f4 + 4];              // K = kbase + 16..19
  const float4 c3 = sT4[f4 + 5];              // K = kbase + 20..23
  const float col[16] = {c0.x, c0.y, c0.z, c0.w, c1.x, c1.y, c1.z, c1.w,
                         c2.x, c2.y, c2.z, c2.w, c3.x, c3.y, c3.z, c3.w};
  v16h av;
#pragma unroll
  for (int e = 0; e < 16; ++e) {
    float arg = (col[e] - ti) * bs;           // -b*(t_i - t_j)
    if (MASKED) {
      const int j = kbase + e + (e & 8);      // CDNA5 16-bit A-fragment K map
      arg = (j < row) ? arg : -__builtin_inff();  // exp(-inf)=0: branchless mask
    }
    av[e] = (_Float16)__expf(arg);
  }
  return __builtin_amdgcn_wmma_f32_16x16x32_f16(
      /*neg_a=*/false, av, /*neg_b=*/false, bones,
      /*c_mod=*/(short)0, acc, /*reuse_a=*/false, /*reuse_b=*/false);
}

// ---------------------------------------------------------------------------
// Kernel 1: product term. One wave32 per 16-row tile, tiles striped across
// blocks for load balance. t_n resident in LDS (32 KB of 320 KB/WGP).
// ---------------------------------------------------------------------------
__global__ __launch_bounds__(32 * WAVES_PER_BLOCK)
void hawkes_product_kernel(const float* __restrict__ t_n,
                           const float* __restrict__ mu,
                           const float* __restrict__ a,
                           const float* __restrict__ b,
                           float* __restrict__ partial_log, int n) {
  __shared__ __align__(16) float sT[N_EVENTS_MAX];
  for (int i = threadIdx.x; i < n; i += blockDim.x) {
    __builtin_prefetch(&t_n[i], 0, 0);  // global_prefetch_b8
    sT[i] = t_n[i];
  }
  __syncthreads();
  const float4* sT4 = reinterpret_cast<const float4*>(sT);

  const int lane = threadIdx.x & 31;
  const int wave = threadIdx.x >> 5;
  const int tile = wave * gridDim.x + blockIdx.x;  // striped for balance
  const int g    = lane >> 4;                      // lane half-group
  const int m    = lane & 15;                      // A-matrix row in tile
  const int row  = tile * TILE_M + m;

  const float ti  = sT[row];
  const float bs  = b[0];
  const float mus = mu[0];
  const float ab  = a[0] * bs;

  v16h bones;
#pragma unroll
  for (int e = 0; e < 16; ++e) bones[e] = (_Float16)1.0f;

  v8f acc = {};

  // Chunks entirely below the diagonal block: no masking needed at all.
  const int kdiag = (tile * TILE_M) & ~(TILE_K - 1);
  const int kend  = (tile + 1) * TILE_M;
  for (int kk = 0; kk < kdiag; kk += TILE_K)
    acc = chunk_step<false>(sT4, kk, g, row, ti, bs, bones, acc);
  // The single chunk straddling the diagonal: branchless j<i mask via -inf arg.
  for (int kk = kdiag; kk < kend; kk += TILE_K)
    acc = chunk_step<true>(sT4, kk, g, row, ti, bs, bones, acc);

  // D layout: lanes 0-15 hold rows 0..7 in acc[0..7]; lanes 16-31 rows 8..15.
  // Every column is identical (B = ones).
  float slog = 0.0f;
#pragma unroll
  for (int r = 0; r < 8; ++r) slog += __logf(mus + ab * acc[r]);

  const float s = __shfl(slog, 0) + __shfl(slog, 16);
  if (lane == 0) partial_log[tile] = s;
}

// ---------------------------------------------------------------------------
// Kernel 2: trapezoid integral of a*b*exp(-b*(t - t_last)) on the 5M grid,
// replicating the reference's numeric sum (incl. the valid_pt[1:] mask).
// Deterministic: fixed thread->segment mapping + fixed-order block tree.
// ---------------------------------------------------------------------------
__global__ __launch_bounds__(INT_THREADS)
void hawkes_integral_kernel(const float* __restrict__ t_n,
                            const float* __restrict__ a,
                            const float* __restrict__ b,
                            float* __restrict__ partial_int, int n) {
  const float t_last = t_n[n - 1];
  const float bs = b[0];
  const float ab = a[0] * bs;

  const int tid = blockIdx.x * blockDim.x + threadIdx.x;
  const int nth = gridDim.x * blockDim.x;

  float s = 0.0f;
  for (int k = tid; k < N_GRID - 1; k += nth) {
    const float g0 = (float)k * TRAP_STEP;
    const float g1 = (float)(k + 1) * TRAP_STEP;
    if (t_last + g1 < T_HORIZON) {
      const float y0 = ab * __expf(-bs * g0);
      const float y1 = ab * __expf(-bs * g1);
      s += 0.5f * (y0 + y1) * TRAP_STEP;
    }
  }

  __shared__ float red[INT_THREADS];
  red[threadIdx.x] = s;
  __syncthreads();
  for (int off = INT_THREADS / 2; off > 0; off >>= 1) {
    if (threadIdx.x < off) red[threadIdx.x] += red[threadIdx.x + off];
    __syncthreads();
  }
  if (threadIdx.x == 0) partial_int[blockIdx.x] = red[0];
}

// ---------------------------------------------------------------------------
// Kernel 3: deterministic serial finalize: out = exp(sum(log terms) - mu*T - I)
// ---------------------------------------------------------------------------
__global__ void hawkes_finalize_kernel(const float* __restrict__ partial_log,
                                       const float* __restrict__ partial_int,
                                       const float* __restrict__ mu,
                                       float* __restrict__ out,
                                       int ntiles, int nint) {
  float lg = 0.0f, iv = 0.0f;
  for (int i = 0; i < ntiles; ++i) lg += partial_log[i];
  for (int i = 0; i < nint; ++i) iv += partial_int[i];
  out[0] = __expf(lg - mu[0] * T_HORIZON - iv);
}

extern "C" void kernel_launch(void* const* d_in, const int* in_sizes, int n_in,
                              void* d_out, int out_size, void* d_ws, size_t ws_size,
                              hipStream_t stream) {
  const float* t_n = (const float*)d_in[0];
  const float* mu  = (const float*)d_in[1];
  const float* a   = (const float*)d_in[2];
  const float* b   = (const float*)d_in[3];
  float* out = (float*)d_out;
  float* ws  = (float*)d_ws;

  const int n      = in_sizes[0];        // 8192
  const int ntiles = n / TILE_M;         // 512

  float* plog = ws;                      // [ntiles]
  float* pint = ws + ntiles;             // [INT_BLOCKS]

  hawkes_product_kernel<<<ntiles / WAVES_PER_BLOCK, 32 * WAVES_PER_BLOCK, 0, stream>>>(
      t_n, mu, a, b, plog, n);
  hawkes_integral_kernel<<<INT_BLOCKS, INT_THREADS, 0, stream>>>(
      t_n, a, b, pint, n);
  hawkes_finalize_kernel<<<1, 1, 0, stream>>>(
      plog, pint, mu, out, ntiles, INT_BLOCKS);
}
